// MultiHeadAttention_1262720385606
// MI455X (gfx1250) — matrix operand order repaired and checked by reading, not compiled, not run on hardware
//
#include <hip/hip_runtime.h>
#include <hip/hip_bf16.h>

// ---------------- problem constants ----------------
#define BB 4
#define TT 2048
#define DD 768
#define HH 12
#define DH 64
// SCALE = sqrt(64) = 8, folded into Q during the QKV GEMM store.

typedef __attribute__((ext_vector_type(16))) __bf16 bhalf16;
typedef __attribute__((ext_vector_type(8)))  __bf16 bhalf8;
typedef __attribute__((ext_vector_type(8)))  float  floatx8;

// A-fragment (16x32 bf16): lane holds row M=lane%16; 16 consecutive K values [k0..k0+15],
// k0 = (lane<16 ? 0 : 16) relative to the 32-wide K chunk.
__device__ __forceinline__ bhalf16 load_frag_a(const __bf16* rowbase, int k0) {
  union { bhalf16 v; bhalf8 h[2]; } u;
  u.h[0] = *(const bhalf8*)(rowbase + k0);
  u.h[1] = *(const bhalf8*)(rowbase + k0 + 8);
  return u.v;
}
// B-fragment (32x16 bf16): lane holds col N=lane%16; 16 consecutive K values starting at
// (lane<16 ? 0 : 16) relative to the 32-wide K chunk -> one contiguous 32B read.
__device__ __forceinline__ bhalf16 load_frag_b(const __bf16* p) {
  union { bhalf16 v; bhalf8 h[2]; } u;
  u.h[0] = ((const bhalf8*)p)[0];
  u.h[1] = ((const bhalf8*)p)[1];
  return u.v;
}
__device__ __forceinline__ floatx8 wmma_bf16(bhalf16 a, bhalf16 b, floatx8 c) {
  return __builtin_amdgcn_wmma_f32_16x16x32_bf16(false, a, false, b, (short)0, c, false, false);
}

// ---- CDNA5 async global->LDS copy (ASYNCcnt-tracked, 16B per lane) ----
// VDST = per-lane LDS byte address (low 32 bits of the generic shared pointer),
// VADDR = per-lane 64-bit global address, no SADDR.
__device__ __forceinline__ void async_ld_b128(const void* gaddr, const __bf16* lds_ptr) {
  unsigned loff = (unsigned)(uintptr_t)lds_ptr;
  asm volatile("global_load_async_to_lds_b128 %0, %1, off"
               :: "v"(loff), "v"(gaddr) : "memory");
}
__device__ __forceinline__ void wait_async_le4() {
  asm volatile("s_wait_asynccnt 0x4" ::: "memory");
}
__device__ __forceinline__ void wait_async_0() {
  asm volatile("s_wait_asynccnt 0x0" ::: "memory");
}

// ---------------- prologue: fp32 -> bf16 conversion / weight packing ----------------
__global__ __launch_bounds__(256) void cvt_x_kernel(const float* __restrict__ x,
                                                    __bf16* __restrict__ xb) {
  int i = (blockIdx.x * 256 + threadIdx.x) * 4;
  float4 f = *(const float4*)(x + i);
  xb[i + 0] = (__bf16)f.x; xb[i + 1] = (__bf16)f.y;
  xb[i + 2] = (__bf16)f.z; xb[i + 3] = (__bf16)f.w;
}

// reference splits e = d*36 + m*12 + h; packed row n = m*768 + h*64 + d so q/k/v columns
// become contiguous [h][d] blocks.
__global__ __launch_bounds__(256) void pack_wqkv_kernel(const float* __restrict__ w,
                                                        __bf16* __restrict__ wb) {
  int i = blockIdx.x * 256 + threadIdx.x;      // 2304*192 threads, 4 K-elems each
  int row = i / 192;                           // packed output row (n index)
  int kk  = (i % 192) * 4;
  int m = row / 768, r = row % 768;
  int h = r >> 6, d = r & 63;
  int srcrow = d * 36 + m * 12 + h;
  float4 f = *(const float4*)(w + (size_t)srcrow * DD + kk);
  __bf16* dst = wb + (size_t)row * DD + kk;
  dst[0] = (__bf16)f.x; dst[1] = (__bf16)f.y; dst[2] = (__bf16)f.z; dst[3] = (__bf16)f.w;
}

__global__ __launch_bounds__(256) void cvt_wout_kernel(const float* __restrict__ w,
                                                       __bf16* __restrict__ wb) {
  int i = (blockIdx.x * 256 + threadIdx.x) * 4;
  float4 f = *(const float4*)(w + i);
  wb[i + 0] = (__bf16)f.x; wb[i + 1] = (__bf16)f.y;
  wb[i + 2] = (__bf16)f.z; wb[i + 3] = (__bf16)f.w;
}

// ---------------- WMMA GEMM: C[M][N] = A[M][K] * W[N][K]^T ----------------
// MODE 0: scatter bf16 into q[B][H][T][DH] (x8), k[B][H][T][DH], v transposed [B][H][DH][T].
// MODE 1: write fp32 to out[M][N].
// Double-buffered async global->LDS staging.
template <int MODE>
__global__ __launch_bounds__(256) void gemm_bf16_wmma(
    const __bf16* __restrict__ A, const __bf16* __restrict__ W,
    __bf16* __restrict__ qb, __bf16* __restrict__ kb, __bf16* __restrict__ vb,
    float* __restrict__ out, int M, int N, int K) {
  __shared__ __bf16 lda[2][128 * 32];
  __shared__ __bf16 ldb[2][128 * 32];
  const int tid = threadIdx.x, lane = tid & 31, wid = tid >> 5;
  const int lr = lane & 15, hs = lane >> 4;
  const int m0 = blockIdx.y * 128, n0 = blockIdx.x * 128;
  const int wm = (wid & 3) * 32;   // 4 waves along M, 2 tiles each
  const int wn = (wid >> 2) * 64;  // 2 waves along N, 4 tiles each
  floatx8 acc[2][4] = {};
  const int ak0 = hs ? 16 : 0;
  const int bk0 = hs ? 16 : 0;

  // per-thread staging slot: one row-half of A and of W (2 x 16B each)
  const int srow = tid >> 1;
  const int sseg = (tid & 1) * 16;
  const __bf16* gA = A + (size_t)(m0 + srow) * K + sseg;
  const __bf16* gB = W + (size_t)(n0 + srow) * K + sseg;
  const int lofs = srow * 32 + sseg;

  const int nK = K / 32;
  // prologue: stage tile 0 into buffer 0
  async_ld_b128(gA + 0, &lda[0][lofs]);
  async_ld_b128(gA + 8, &lda[0][lofs + 8]);
  async_ld_b128(gB + 0, &ldb[0][lofs]);
  async_ld_b128(gB + 8, &ldb[0][lofs + 8]);

  int cur = 0;
  for (int kblk = 0; kblk < nK; ++kblk) {
    if (kblk + 1 < nK) {  // issue next tile, then wait for current tile (in-order)
      int ks = (kblk + 1) * 32;
      async_ld_b128(gA + ks,     &lda[cur ^ 1][lofs]);
      async_ld_b128(gA + ks + 8, &lda[cur ^ 1][lofs + 8]);
      async_ld_b128(gB + ks,     &ldb[cur ^ 1][lofs]);
      async_ld_b128(gB + ks + 8, &ldb[cur ^ 1][lofs + 8]);
      wait_async_le4();
    } else {
      wait_async_0();
    }
    __syncthreads();
    bhalf16 af[2], bfr[4];
#pragma unroll
    for (int mt = 0; mt < 2; ++mt)
      af[mt] = load_frag_a(&lda[cur][(wm + mt * 16 + lr) * 32], ak0);
#pragma unroll
    for (int nt = 0; nt < 4; ++nt)
      bfr[nt] = load_frag_b(&ldb[cur][(wn + nt * 16 + lr) * 32 + bk0]);
#pragma unroll
    for (int mt = 0; mt < 2; ++mt)
#pragma unroll
      for (int nt = 0; nt < 4; ++nt)
        acc[mt][nt] = wmma_bf16(af[mt], bfr[nt], acc[mt][nt]);
    __syncthreads();
    cur ^= 1;
  }

  // stores: tiles are 16-aligned and never straddle 768/2048 boundaries -> hoist index math
#pragma unroll
  for (int mt = 0; mt < 2; ++mt) {
    const int row0 = m0 + wm + mt * 16 + 8 * hs;  // C layout: M = vgpr + 8*(lane>=16)
#pragma unroll
    for (int nt = 0; nt < 4; ++nt) {
      const int col0 = n0 + wn + nt * 16;
      if (MODE == 0) {
        const int m = col0 / 768, rr = col0 % 768;
        const int h = rr >> 6, d0 = (rr & 63) + lr;
        const int b = row0 >> 11, t0 = row0 & 2047;
        if (m == 0) {
          __bf16* dst = qb + (((size_t)(b * HH + h)) * TT + t0) * DH + d0;
#pragma unroll
          for (int r = 0; r < 8; ++r) dst[(size_t)r * DH] = (__bf16)(acc[mt][nt][r] * 8.0f);
        } else if (m == 1) {
          __bf16* dst = kb + (((size_t)(b * HH + h)) * TT + t0) * DH + d0;
#pragma unroll
          for (int r = 0; r < 8; ++r) dst[(size_t)r * DH] = (__bf16)acc[mt][nt][r];
        } else {  // V stored transposed: [b][h][d][t]
          __bf16* dst = vb + (((size_t)(b * HH + h)) * DH + d0) * TT + t0;
#pragma unroll
          for (int r = 0; r < 8; ++r) dst[r] = (__bf16)acc[mt][nt][r];
        }
      } else {
        float* dst = out + (size_t)row0 * N + col0 + lr;
#pragma unroll
        for (int r = 0; r < 8; ++r) dst[(size_t)r * N] = acc[mt][nt][r];
      }
    }
  }
}

// ---------------- flash attention: per (b,h), 128 query rows / block ----------------
__global__ __launch_bounds__(256) void flash_attn_kernel(
    const __bf16* __restrict__ qg, const __bf16* __restrict__ kg,
    const __bf16* __restrict__ vg, __bf16* __restrict__ ob) {
  __shared__ __bf16 ldk[2][64 * 64];   // K tile  [key][feat]
  __shared__ __bf16 ldvt[2][64 * 64];  // V^T tile [feat][key] (direct copy: v is [b][h][d][t])
  __shared__ __bf16 ldp[8][16 * 64];   // per-wave P staging (C-layout -> A-layout)
  const int tid = threadIdx.x, lane = tid & 31, wid = tid >> 5;
  const int lr = lane & 15, hs = lane >> 4;
  const int b = blockIdx.z, h = blockIdx.y;
  const int q0 = blockIdx.x * 128 + wid * 16;
  const size_t head = ((size_t)(b * HH + h)) * TT * DH;
  const __bf16* Q = qg + head;
  const __bf16* Kp = kg + head;
  const __bf16* Vp = vg + head;  // transposed layout, same flat extent
  const int ak0 = hs ? 16 : 0;
  const int bk0 = hs ? 16 : 0;

  // Q fragments held in registers for the whole pass (16 rows x 64 feats)
  bhalf16 qf[2];
#pragma unroll
  for (int c = 0; c < 2; ++c) qf[c] = load_frag_a(Q + (size_t)(q0 + lr) * DH, c * 32 + ak0);

  float mrun[8], lrun[8];
  floatx8 acc[4] = {};
#pragma unroll
  for (int r = 0; r < 8; ++r) { mrun[r] = -3.0e38f; lrun[r] = 0.0f; }

  // per-thread staging slot: row-quarter of K tile and of V^T tile
  const int srow = tid >> 2;           // 0..63: key (for K) / feat (for V^T)
  const int sseg = (tid & 3) * 16;     // 16-half segment
  const __bf16* gK = Kp + (size_t)srow * DH + sseg;   // advance by kb0*DH... rows are keys
  const __bf16* gV = Vp + (size_t)srow * TT + sseg;   // advance by kb0 along keys
  const int lofs = srow * 64 + sseg;

  // prologue: stage key-block 0
  async_ld_b128(gK + 0, &ldk[0][lofs]);
  async_ld_b128(gK + 8, &ldk[0][lofs + 8]);
  async_ld_b128(gV + 0, &ldvt[0][lofs]);
  async_ld_b128(gV + 8, &ldvt[0][lofs + 8]);

  int cur = 0;
  for (int kb0 = 0; kb0 < TT; kb0 += 64) {
    if (kb0 + 64 < TT) {
      const __bf16* nK = gK + (size_t)(kb0 + 64) * DH;
      const __bf16* nV = gV + (kb0 + 64);
      async_ld_b128(nK + 0, &ldk[cur ^ 1][lofs]);
      async_ld_b128(nK + 8, &ldk[cur ^ 1][lofs + 8]);
      async_ld_b128(nV + 0, &ldvt[cur ^ 1][lofs]);
      async_ld_b128(nV + 8, &ldvt[cur ^ 1][lofs + 8]);
      wait_async_le4();
    } else {
      wait_async_0();
    }
    __syncthreads();

    // S = Q K^T (16 q-rows x 64 keys), already scaled via pre-scaled Q
    floatx8 s[4];
#pragma unroll
    for (int nt = 0; nt < 4; ++nt) {
      floatx8 z = {};
      bhalf16 bk0f = load_frag_b(&ldk[cur][(nt * 16 + lr) * 64 + 0 + bk0]);
      bhalf16 bk1f = load_frag_b(&ldk[cur][(nt * 16 + lr) * 64 + 32 + bk0]);
      z = wmma_bf16(qf[0], bk0f, z);
      z = wmma_bf16(qf[1], bk1f, z);
      s[nt] = z;
    }

    // online softmax (row = r + 8*hs; 16-lane shfl_xor reductions stay in-half)
#pragma unroll
    for (int r = 0; r < 8; ++r) {
      float bm = fmaxf(fmaxf(s[0][r], s[1][r]), fmaxf(s[2][r], s[3][r]));
      bm = fmaxf(bm, __shfl_xor(bm, 1, 32));
      bm = fmaxf(bm, __shfl_xor(bm, 2, 32));
      bm = fmaxf(bm, __shfl_xor(bm, 4, 32));
      bm = fmaxf(bm, __shfl_xor(bm, 8, 32));
      float mn = fmaxf(mrun[r], bm);
      float al = __expf(mrun[r] - mn);
      float rs = 0.0f;
#pragma unroll
      for (int nt = 0; nt < 4; ++nt) {
        float p = __expf(s[nt][r] - mn);
        s[nt][r] = p;
        rs += p;
      }
      rs += __shfl_xor(rs, 1, 32);
      rs += __shfl_xor(rs, 2, 32);
      rs += __shfl_xor(rs, 4, 32);
      rs += __shfl_xor(rs, 8, 32);
      lrun[r] = lrun[r] * al + rs;
      mrun[r] = mn;
#pragma unroll
      for (int nt = 0; nt < 4; ++nt) acc[nt][r] *= al;
    }

    // C-layout P -> per-wave LDS -> A-layout fragments (wave-local, dscnt-guarded)
    __bf16* pw = &ldp[wid][0];
#pragma unroll
    for (int nt = 0; nt < 4; ++nt)
#pragma unroll
      for (int r = 0; r < 8; ++r)
        pw[(r + 8 * hs) * 64 + nt * 16 + lr] = (__bf16)s[nt][r];
    asm volatile("s_wait_dscnt 0x0" ::: "memory");
    bhalf16 pf[2];
#pragma unroll
    for (int c = 0; c < 2; ++c) pf[c] = load_frag_a(pw + (size_t)lr * 64, c * 32 + ak0);

    // O += P V  (keys are the K dim; V^T tile gives contiguous B-fragments)
#pragma unroll
    for (int ft = 0; ft < 4; ++ft) {
      bhalf16 bv0 = load_frag_b(&ldvt[cur][(ft * 16 + lr) * 64 + 0 + bk0]);
      bhalf16 bv1 = load_frag_b(&ldvt[cur][(ft * 16 + lr) * 64 + 32 + bk0]);
      acc[ft] = wmma_bf16(pf[0], bv0, acc[ft]);
      acc[ft] = wmma_bf16(pf[1], bv1, acc[ft]);
    }
    __syncthreads();
    cur ^= 1;
  }

  // normalize and store o in [b][t][(h d)] bf16 layout for the output GEMM
#pragma unroll
  for (int r = 0; r < 8; ++r) {
    float inv = 1.0f / lrun[r];
    int t = q0 + r + 8 * hs;
    __bf16* dst = ob + ((size_t)(b * TT + t)) * DD + h * DH;
#pragma unroll
    for (int ft = 0; ft < 4; ++ft) dst[ft * 16 + lr] = (__bf16)(acc[ft][r] * inv);
  }
}

// ---------------- host launch ----------------
extern "C" void kernel_launch(void* const* d_in, const int* in_sizes, int n_in,
                              void* d_out, int out_size, void* d_ws, size_t ws_size,
                              hipStream_t stream) {
  const float* x     = (const float*)d_in[0];  // [4,2048,768]
  const float* w_qkv = (const float*)d_in[1];  // [2304,768]
  const float* w_out = (const float*)d_in[2];  // [768,768]
  float* out = (float*)d_out;                  // [4,2048,768]

  char* ws = (char*)d_ws;
  const size_t SZ_XB = (size_t)BB * TT * DD * 2;      // 12,582,912
  const size_t SZ_WQ = (size_t)3 * DD * DD * 2;       //  3,538,944
  const size_t SZ_WO = (size_t)DD * DD * 2;           //  1,179,648
  const size_t SZ_H  = (size_t)BB * HH * TT * DH * 2; // 12,582,912
  __bf16* xb    = (__bf16*)(ws);
  __bf16* wqkvb = (__bf16*)(ws + SZ_XB);
  __bf16* woutb = (__bf16*)(ws + SZ_XB + SZ_WQ);
  __bf16* qb    = (__bf16*)(ws + SZ_XB + SZ_WQ + SZ_WO);
  __bf16* kb    = (__bf16*)(ws + SZ_XB + SZ_WQ + SZ_WO + SZ_H);
  __bf16* vb    = (__bf16*)(ws + SZ_XB + SZ_WQ + SZ_WO + 2 * SZ_H);
  __bf16* ob    = (__bf16*)(ws + SZ_XB + SZ_WQ + SZ_WO + 3 * SZ_H);
  // total workspace use: ~67.6 MB

  cvt_x_kernel<<<(BB * TT * DD / 4) / 256, 256, 0, stream>>>(x, xb);
  pack_wqkv_kernel<<<(3 * DD * DD / 4) / 256, 256, 0, stream>>>(w_qkv, wqkvb);
  cvt_wout_kernel<<<(DD * DD / 4) / 256, 256, 0, stream>>>(w_out, woutb);

  // QKV projection: M=8192, N=2304, K=768 (scatters q/k, v transposed; scales q by 8)
  gemm_bf16_wmma<0><<<dim3(2304 / 128, 8192 / 128), 256, 0, stream>>>(
      xb, wqkvb, qb, kb, vb, nullptr, BB * TT, 3 * DD, DD);

  // attention: grid (T/128, H, B)
  flash_attn_kernel<<<dim3(TT / 128, HH, BB), 256, 0, stream>>>(qb, kb, vb, ob);

  // output projection: M=8192, N=768, K=768, fp32 out
  gemm_bf16_wmma<1><<<dim3(768 / 128, 8192 / 128), 256, 0, stream>>>(
      ob, woutb, nullptr, nullptr, nullptr, out, BB * TT, DD, DD);
}
